// TiSASRec_36636071035272
// MI455X (gfx1250) — matrix operand order repaired and checked by reading, not compiled, not run on hardware
//
#include <hip/hip_runtime.h>
#include <hip/hip_bf16.h>

#define B_ 8
#define L_ 256
#define H_ 256
#define HEADS_ 4
#define D_ 64
#define NBLK_ 2
#define M_ (B_ * L_)            // 2048 rows
#define BIG_NEG (-4294967295.0f)

typedef __attribute__((ext_vector_type(16))) _Float16 v16h;
typedef __attribute__((ext_vector_type(8)))  float    v8f;

__device__ __forceinline__ float4 f4add(float4 a, float4 b) {
    return make_float4(a.x + b.x, a.y + b.y, a.z + b.z, a.w + b.w);
}

// pack 16 f32 (4x float4) -> v16h
__device__ __forceinline__ v16h pack16(float4 a, float4 b, float4 c, float4 d) {
    v16h v;
    v[0]  = (_Float16)a.x; v[1]  = (_Float16)a.y; v[2]  = (_Float16)a.z; v[3]  = (_Float16)a.w;
    v[4]  = (_Float16)b.x; v[5]  = (_Float16)b.y; v[6]  = (_Float16)b.z; v[7]  = (_Float16)b.w;
    v[8]  = (_Float16)c.x; v[9]  = (_Float16)c.y; v[10] = (_Float16)c.z; v[11] = (_Float16)c.w;
    v[12] = (_Float16)d.x; v[13] = (_Float16)d.y; v[14] = (_Float16)d.z; v[15] = (_Float16)d.w;
    return v;
}

// Load CDNA5 16-bit A-fragment (16x32, M x K) for this lane from f32 row.
// Lane l: row = ln (l&15), hi = l>>4.  e -> k = hi*16+e (16 contiguous K values)
__device__ __forceinline__ v16h loadA(const float* __restrict__ row, int hi) {
    const float4* p0 = (const float4*)(row + hi * 16);
    const float4* p1 = (const float4*)(row + hi * 16 + 8);
    return pack16(p0[0], p0[1], p1[0], p1[1]);
}

// Load CDNA5 16-bit B-fragment (32x16, K x N): lane holds col n=l&15,
// e -> k = hi*16 + e (16 contiguous K values).
__device__ __forceinline__ v16h loadB(const float* __restrict__ colrun) {
    const float4* p = (const float4*)colrun;
    return pack16(p[0], p[1], p[2], p[3]);
}

__device__ __forceinline__ v16h loadB_sum(const float* __restrict__ c0,
                                          const float* __restrict__ c1) {
    const float4* p = (const float4*)c0;
    const float4* q = (const float4*)c1;
    return pack16(f4add(p[0], q[0]), f4add(p[1], q[1]),
                  f4add(p[2], q[2]), f4add(p[3], q[3]));
}

// ---------------------------------------------------------------------------
// x = seq_emb * keep
// ---------------------------------------------------------------------------
__global__ __launch_bounds__(256) void prep_x(const float* __restrict__ seq,
                                              const int* __restrict__ mask,
                                              float* __restrict__ x) {
    int i = blockIdx.x * 256 + threadIdx.x;
    int row = i / H_;
    x[i] = seq[i] * (mask[row] ? 0.0f : 1.0f);
}

// ---------------------------------------------------------------------------
// out[row,:] = LN(in[row,:]) * g + b    (H == blockDim == 256)
// ---------------------------------------------------------------------------
__global__ __launch_bounds__(256) void layernorm_k(const float* __restrict__ in,
                                                   const float* __restrict__ g,
                                                   const float* __restrict__ b,
                                                   float* __restrict__ out) {
    __shared__ float red[256];
    int row = blockIdx.x, t = threadIdx.x;
    size_t idx = (size_t)row * H_ + t;
    float v = in[idx];
    red[t] = v; __syncthreads();
    for (int s = 128; s > 0; s >>= 1) { if (t < s) red[t] += red[t + s]; __syncthreads(); }
    float m = red[0] * (1.0f / H_); __syncthreads();
    float d = v - m;
    red[t] = d * d; __syncthreads();
    for (int s = 128; s > 0; s >>= 1) { if (t < s) red[t] += red[t + s]; __syncthreads(); }
    float var = red[0] * (1.0f / H_);
    out[idx] = d * rsqrtf(var + 1e-8f) * g[t] + b[t];
}

// ---------------------------------------------------------------------------
// Y[2048,256] = act(X @ W^T + bias) (+resid) (*keep)    WMMA f16->f32
// ---------------------------------------------------------------------------
__global__ __launch_bounds__(256) void gemm_wT_wmma(
        const float* __restrict__ X, const float* __restrict__ W,
        const float* __restrict__ bias, const float* __restrict__ resid,
        const int* __restrict__ mask, float* __restrict__ Y, int relu) {
    int wave = threadIdx.x >> 5, lane = threadIdx.x & 31;
    int hi = lane >> 4, ln = lane & 15;
    int gtile = blockIdx.x * 8 + wave;
    int tm0 = (gtile >> 4) * 16;           // 16 tiles along N (256/16)
    int tn0 = (gtile & 15) * 16;

    const float* xrow = X + (size_t)(tm0 + ln) * H_;
    const float* wrow = W + (size_t)(tn0 + ln) * H_ + hi * 16;
    v8f acc = {};
    #pragma unroll 2
    for (int kk = 0; kk < H_; kk += 32) {
        v16h a  = loadA(xrow + kk, hi);
        v16h bm = loadB(wrow + kk);
        acc = __builtin_amdgcn_wmma_f32_16x16x32_f16(false, a, false, bm,
                                                     (short)0, acc, false, false);
    }
    int col = tn0 + ln;
    float bv = bias[col];
    #pragma unroll
    for (int r = 0; r < 8; ++r) {
        int row = tm0 + r + hi * 8;
        float v = acc[r] + bv;
        if (relu) v = fmaxf(v, 0.0f);
        if (resid) v += resid[(size_t)row * H_ + col];
        if (mask)  v *= (mask[row] ? 0.0f : 1.0f);
        Y[(size_t)row * H_ + col] = v;
    }
}

// ---------------------------------------------------------------------------
// aw[b,h,q,k] = (Q.(K+posK) [WMMA] + Q.tK_emb[t[b,q,k]] [VALU]) / sqrt(D)
// query-pad + causal masks.  Grid: (L/16, HEADS, B), 8 waves.
// ---------------------------------------------------------------------------
__global__ __launch_bounds__(256) void attn_scores(
        const float* __restrict__ Q, const float* __restrict__ Kh,
        const float* __restrict__ posK, const float* __restrict__ tK,
        const int* __restrict__ tmat, const int* __restrict__ mask,
        float* __restrict__ aw) {
    int b = blockIdx.z, h = blockIdx.y, q0 = blockIdx.x * 16;
    int wave = threadIdx.x >> 5, lane = threadIdx.x & 31;
    int hi = lane >> 4, ln = lane & 15;
    const float inv_scale = 0.125f;                    // 1/sqrt(64)

    const float* qrowA = Q + (size_t)(b * L_ + q0 + ln) * H_ + h * D_;

    for (int kt = wave; kt < 16; kt += 8) {
        int k0 = kt * 16;
        int kcol = k0 + ln;
        const float* krow = Kh   + (size_t)(b * L_ + kcol) * H_ + h * D_ + hi * 16;
        const float* prow = posK + (size_t)kcol * H_ + h * D_ + hi * 16;
        v8f acc = {};
        #pragma unroll
        for (int dd = 0; dd < D_; dd += 32) {
            v16h a  = loadA(qrowA + dd, hi);
            v16h bm = loadB_sum(krow + dd, prow + dd);
            acc = __builtin_amdgcn_wmma_f32_16x16x32_f16(false, a, false, bm,
                                                         (short)0, acc, false, false);
        }
        int kg = k0 + ln;
        #pragma unroll
        for (int r = 0; r < 8; ++r) {
            int q = q0 + r + hi * 8;
            int t = tmat[(size_t)(b * L_ + q) * L_ + kg];
            const float4* q4 = (const float4*)(Q + (size_t)(b * L_ + q) * H_ + h * D_);
            const float4* e4 = (const float4*)(tK + (size_t)t * H_ + h * D_);
            float s = 0.0f;
            #pragma unroll 4
            for (int d = 0; d < D_ / 4; ++d) {
                float4 qv = q4[d], ev = e4[d];
                s += qv.x * ev.x + qv.y * ev.y + qv.z * ev.z + qv.w * ev.w;
            }
            float v = (acc[r] + s) * inv_scale;
            if (mask[b * L_ + q] || kg > q) v = BIG_NEG;   // query-pad | causal
            aw[((size_t)(b * HEADS_ + h) * L_ + q) * L_ + kg] = v;
        }
    }
}

// ---------------------------------------------------------------------------
// In-place row softmax over k (row width 256).
// ---------------------------------------------------------------------------
__global__ __launch_bounds__(256) void softmax_k(float* __restrict__ p) {
    __shared__ float red[256];
    size_t row = blockIdx.x;
    int t = threadIdx.x;
    float v = p[row * L_ + t];
    red[t] = v; __syncthreads();
    for (int s = 128; s > 0; s >>= 1) { if (t < s) red[t] = fmaxf(red[t], red[t + s]); __syncthreads(); }
    float mx = red[0]; __syncthreads();
    float e = __expf(v - mx);
    red[t] = e; __syncthreads();
    for (int s = 128; s > 0; s >>= 1) { if (t < s) red[t] += red[t + s]; __syncthreads(); }
    p[row * L_ + t] = e / red[0];
}

// ---------------------------------------------------------------------------
// Yo[b,q,h,d] = sum_k P[b,h,q,k] * (V[b,k,h,d] + posV[k,h,d])   (WMMA)
// Grid: (B*HEADS, L/32); wave -> (q-subtile, d-tile).
// ---------------------------------------------------------------------------
__global__ __launch_bounds__(256) void attn_out_wmma(
        const float* __restrict__ P, const float* __restrict__ V,
        const float* __restrict__ posV, float* __restrict__ Yo) {
    int bh = blockIdx.x, b = bh / HEADS_, h = bh % HEADS_;
    int wave = threadIdx.x >> 5, lane = threadIdx.x & 31;
    int hi = lane >> 4, ln = lane & 15;
    int q0 = blockIdx.y * 32 + (wave >> 2) * 16;
    int d0 = (wave & 3) * 16;
    const float* prow = P + ((size_t)bh * L_ + q0 + ln) * L_;
    int d = d0 + ln;

    v8f acc = {};
    #pragma unroll 2
    for (int kk = 0; kk < L_; kk += 32) {
        v16h a = loadA(prow + kk, hi);
        v16h bm;
        #pragma unroll
        for (int e = 0; e < 16; ++e) {
            int k = kk + hi * 16 + e;
            bm[e] = (_Float16)(V[(size_t)(b * L_ + k) * H_ + h * D_ + d] +
                               posV[(size_t)k * H_ + h * D_ + d]);
        }
        acc = __builtin_amdgcn_wmma_f32_16x16x32_f16(false, a, false, bm,
                                                     (short)0, acc, false, false);
    }
    #pragma unroll
    for (int r = 0; r < 8; ++r)
        Yo[(size_t)(b * L_ + q0 + r + hi * 8) * H_ + h * D_ + d] = acc[r];
}

// ---------------------------------------------------------------------------
// out[b,q,c] = qn + Yo + sum_k P[b,h(c),q,k] * tV_emb[t[b,q,k]][c]
// Grid: (L, B), thread = column c in H.
// ---------------------------------------------------------------------------
__global__ __launch_bounds__(256) void attn_tmv_resid(
        const float* __restrict__ P, const float* __restrict__ tV,
        const int* __restrict__ tmat, const float* __restrict__ qn,
        const float* __restrict__ yw, float* __restrict__ out) {
    int q = blockIdx.x, b = blockIdx.y, c = threadIdx.x;
    int h = c >> 6;
    const float* Pb = P + ((size_t)(b * HEADS_ + h) * L_ + q) * L_;
    const int* tr = tmat + (size_t)(b * L_ + q) * L_;
    float acc = 0.0f;
    #pragma unroll 4
    for (int k = 0; k < L_; ++k) {
        int t = tr[k];
        acc += Pb[k] * tV[(size_t)t * H_ + c];
    }
    size_t idx = (size_t)(b * L_ + q) * H_ + c;
    out[idx] = qn[idx] + yw[idx] + acc;
}

// ---------------------------------------------------------------------------
extern "C" void kernel_launch(void* const* d_in, const int* in_sizes, int n_in,
                              void* d_out, int out_size, void* d_ws, size_t ws_size,
                              hipStream_t stream) {
    const int*   mask = (const int*)d_in[0];
    const float* seq  = (const float*)d_in[1];
    const int*   tmat = (const int*)d_in[3];
    const float* Wq   = (const float*)d_in[5];
    const float* bq   = (const float*)d_in[6];
    const float* Wk   = (const float*)d_in[7];
    const float* bk   = (const float*)d_in[8];
    const float* Wv   = (const float*)d_in[9];
    const float* bv   = (const float*)d_in[10];
    const float* ln1g = (const float*)d_in[11];
    const float* ln1b = (const float*)d_in[12];
    const float* ln2g = (const float*)d_in[13];
    const float* ln2b = (const float*)d_in[14];
    const float* W1   = (const float*)d_in[15];
    const float* b1   = (const float*)d_in[16];
    const float* W2   = (const float*)d_in[17];
    const float* b2   = (const float*)d_in[18];
    const float* posK = (const float*)d_in[19];
    const float* posV = (const float*)d_in[20];
    const float* tK   = (const float*)d_in[21];
    const float* tV   = (const float*)d_in[22];
    const float* lnfg = (const float*)d_in[23];
    const float* lnfb = (const float*)d_in[24];

    const size_t ACT = (size_t)M_ * H_;    // 524288 floats
    float* ws = (float*)d_ws;
    float* x   = ws;                       // activations
    float* qn  = x  + ACT;                 // LN1 output
    float* Qh  = qn + ACT;                 // Q  (aliased as FFN hidden)
    float* Kh  = Qh + ACT;                 // K  (aliased as pre-LN2 temp)
    float* Vh  = Kh + ACT;                 // V
    float* yw  = Vh + ACT;                 // WMMA attn-out
    float* aw  = yw + ACT;                 // [B,HEADS,L,L] scores/probs
    float* hb  = Qh;                       // FFN hidden  (alias: Qh dead after scores)
    float* tmp = Kh;                       // pre-LN2 sum (alias: Kh dead after scores)

    const int gemm_blocks = (M_ / 16) * (H_ / 16) / 8;   // 256

    prep_x<<<M_, 256, 0, stream>>>(seq, mask, x);

    for (int i = 0; i < NBLK_; ++i) {
        const float* wq = Wq + (size_t)i * H_ * H_;
        const float* wk = Wk + (size_t)i * H_ * H_;
        const float* wv = Wv + (size_t)i * H_ * H_;
        const float* w1 = W1 + (size_t)i * H_ * H_;
        const float* w2 = W2 + (size_t)i * H_ * H_;

        layernorm_k<<<M_, 256, 0, stream>>>(x, ln1g + i * H_, ln1b + i * H_, qn);

        gemm_wT_wmma<<<gemm_blocks, 256, 0, stream>>>(qn, wq, bq + i * H_, nullptr, nullptr, Qh, 0);
        gemm_wT_wmma<<<gemm_blocks, 256, 0, stream>>>(x,  wk, bk + i * H_, nullptr, nullptr, Kh, 0);
        gemm_wT_wmma<<<gemm_blocks, 256, 0, stream>>>(x,  wv, bv + i * H_, nullptr, nullptr, Vh, 0);

        attn_scores<<<dim3(L_ / 16, HEADS_, B_), 256, 0, stream>>>(Qh, Kh, posK, tK, tmat, mask, aw);
        softmax_k<<<B_ * HEADS_ * L_, 256, 0, stream>>>(aw);
        attn_out_wmma<<<dim3(B_ * HEADS_, L_ / 32), 256, 0, stream>>>(aw, Vh, posV, yw);
        attn_tmv_resid<<<dim3(L_, B_), 256, 0, stream>>>(aw, tV, tmat, qn, yw, tmp);

        layernorm_k<<<M_, 256, 0, stream>>>(tmp, ln2g + i * H_, ln2b + i * H_, x);

        gemm_wT_wmma<<<gemm_blocks, 256, 0, stream>>>(x,  w1, b1 + i * H_, nullptr, nullptr, hb, 1);
        gemm_wT_wmma<<<gemm_blocks, 256, 0, stream>>>(hb, w2, b2 + i * H_, x, mask, x, 0);
    }

    layernorm_k<<<M_, 256, 0, stream>>>(x, lnfg, lnfb, (float*)d_out);
}
